// Attention_49460843381603
// MI455X (gfx1250) — matrix operand order repaired and checked by reading, not compiled, not run on hardware
//
#include <hip/hip_runtime.h>

#define BATCH 4
#define DIM   256
#define LEN   4096
#define QSTR  264   // padded LDS row stride (bf16 elems) for projected Q
#define PBSTR 40    // padded LDS row stride (bf16 elems) for staged P

typedef __attribute__((ext_vector_type(16))) __bf16      v16bf;
typedef __attribute__((ext_vector_type(8)))  float        v8f;
typedef __attribute__((ext_vector_type(4)))  unsigned int u32x4;

union FragU { u32x4 u[2]; v16bf v; };

// Load a 16-element bf16 fragment from two 16-byte chunks.
static __device__ inline v16bf load_frag2(const __bf16* p0, const __bf16* p1) {
  FragU f;
  f.u[0] = *reinterpret_cast<const u32x4*>(p0);
  f.u[1] = *reinterpret_cast<const u32x4*>(p1);
  return f.v;
}

static __device__ inline v8f wmma_bf16(v16bf a, v16bf b, v8f c) {
  // D = A(16x32) * B(32x16) + C, f32 accumulate
  return __builtin_amdgcn_wmma_f32_16x16x32_bf16(false, a, false, b, (short)0, c,
                                                 false, false);
}

// ---------------------------------------------------------------------------
// Prep: all f32 -> bf16 conversion happens here, so the hot kernel only does
// contiguous 16B bf16 fragment loads + WMMA.
//   ctxb[b][d][q]          natural layout; AV B-fragments contiguous per lane
//   ctxP[b][kc][q][d&31]   d-chunk swizzle; QK^T B-fragments contiguous
//   qT  [b][o][k]          transposed query; projection A-fragments contiguous
//   Wb  [d][k]             bf16 W; projection B-fragments contiguous
// ---------------------------------------------------------------------------
__global__ __launch_bounds__(256) void prep_all(
    const float* __restrict__ ctx, const float* __restrict__ query,
    const float* __restrict__ W, __bf16* __restrict__ ctxb,
    __bf16* __restrict__ ctxP, __bf16* __restrict__ qT,
    __bf16* __restrict__ Wb) {
  size_t idx = (size_t)blockIdx.x * 256 + threadIdx.x;
  unsigned q = idx & (LEN - 1);          // inner position (o for query)
  unsigned d = (idx >> 12) & (DIM - 1);  // middle (k for query)
  unsigned b = idx >> 20;

  float cv  = ctx[idx];
  __bf16 cb = (__bf16)cv;
  ctxb[idx] = cb;
  ctxP[(((size_t)(b * 8 + (d >> 5)) * LEN) + q) * 32 + (d & 31)] = cb;

  float qv = query[idx];
  qT[((size_t)b * LEN + q) * DIM + d] = (__bf16)qv;

  if (idx < DIM * DIM) Wb[idx] = (__bf16)W[idx];
}

// ---------------------------------------------------------------------------
// Fused attention: per workgroup = one batch b, 64 output rows (4 row groups).
//   Phase 0: project Q tile with WMMA into LDS (bf16), bias in accumulator.
//   Phase 1: row sums of exp(scores)   (WMMA + lane butterfly, deterministic)
//   Phase 2: recompute scores, write normalized weights, accumulate AV (WMMA)
// ---------------------------------------------------------------------------
__global__ __launch_bounds__(256) void attn_kernel(
    const __bf16* __restrict__ qT, const __bf16* __restrict__ Wb,
    const float* __restrict__ bias, const __bf16* __restrict__ ctxb,
    const __bf16* __restrict__ ctxP, float* __restrict__ out,
    float* __restrict__ weights) {
  __shared__ __bf16 qpb[64 * QSTR];        // projected Q, bf16 [row][d]
  __shared__ __bf16 pbuf[64 * PBSTR];      // staged softmax probs, [row][32 q]
  __shared__ float  wsum[8][64];           // per-wave partial row sums
  __shared__ float  stats[64];             // 1 / rowsum

  const int tid  = threadIdx.x;
  const int lane = tid & 31;
  const int wave = tid >> 5;
  const int nlo  = lane & 15;
  const int hb   = lane >> 4;              // half-wave select

  const int b  = blockIdx.x >> 6;          // 64 row-blocks per batch
  const int o0 = (blockIdx.x & 63) * 64;

  const v8f vzero = {0.f, 0.f, 0.f, 0.f, 0.f, 0.f, 0.f, 0.f};

  // ---------------- Phase 0: projection into qpb (LDS) --------------------
  {
    const int rg = wave >> 1;              // this wave's row group
    v16bf afr[8];
    const __bf16* qp = qT + ((size_t)b * LEN + o0 + rg * 16 + nlo) * DIM;
    #pragma unroll
    for (int kc = 0; kc < 8; ++kc) {
      const __bf16* ap = qp + kc * 32 + hb * 8;
      afr[kc] = load_frag2(ap, ap + 16);
    }
    #pragma unroll 2
    for (int j = 0; j < 8; ++j) {
      const int d0 = ((wave & 1) * 8 + j) * 16;
      const float bv = bias[d0 + nlo];
      v8f acc = {bv, bv, bv, bv, bv, bv, bv, bv};
      const __bf16* wp0 = Wb + (size_t)(d0 + nlo) * DIM + hb * 8;
      #pragma unroll
      for (int kc = 0; kc < 8; ++kc) {
        const __bf16* wp = wp0 + kc * 32;
        v16bf bf = load_frag2(wp, wp + 16);
        acc = wmma_bf16(afr[kc], bf, acc);
      }
      #pragma unroll
      for (int r = 0; r < 8; ++r)
        qpb[(rg * 16 + hb * 8 + r) * QSTR + d0 + nlo] = (__bf16)acc[r];
    }
  }
  __syncthreads();

  // ---------------- Phase 1: row sums of exp(scores) ----------------------
  {
    const __bf16* ctxPb = ctxP + (size_t)b * 8 * LEN * 32;
    float vsum[4][8];
    #pragma unroll
    for (int rg = 0; rg < 4; ++rg)
      #pragma unroll
      for (int r = 0; r < 8; ++r) vsum[rg][r] = 0.f;

    for (int jt = wave; jt < 256; jt += 8) {
      const int q0 = jt * 16;
      v8f sacc[4];
      #pragma unroll
      for (int rg = 0; rg < 4; ++rg) sacc[rg] = vzero;
      #pragma unroll
      for (int kc = 0; kc < 8; ++kc) {
        const __bf16* bp = ctxPb + ((size_t)kc * LEN + q0 + nlo) * 32 + hb * 8;
        v16bf bf = load_frag2(bp, bp + 16);       // B shared by 4 row groups
        #pragma unroll
        for (int rg = 0; rg < 4; ++rg) {
          const __bf16* ap = qpb + (rg * 16 + nlo) * QSTR + kc * 32 + hb * 8;
          v16bf af = load_frag2(ap, ap + 16);
          sacc[rg] = wmma_bf16(af, bf, sacc[rg]);
        }
      }
      #pragma unroll
      for (int rg = 0; rg < 4; ++rg)
        #pragma unroll
        for (int r = 0; r < 8; ++r) {
          float p = __expf(sacc[rg][r]);
          p += __shfl_xor(p, 1);
          p += __shfl_xor(p, 2);
          p += __shfl_xor(p, 4);
          p += __shfl_xor(p, 8);                  // reduce 16 cols, stay in half
          vsum[rg][r] += p;
        }
    }
    if (nlo == 0) {
      #pragma unroll
      for (int rg = 0; rg < 4; ++rg)
        #pragma unroll
        for (int r = 0; r < 8; ++r)
          wsum[wave][rg * 16 + hb * 8 + r] = vsum[rg][r];
    }
  }
  __syncthreads();
  if (tid < 64) {
    float s = 0.f;
    #pragma unroll
    for (int w = 0; w < 8; ++w) s += wsum[w][tid];   // fixed order: deterministic
    stats[tid] = 1.0f / s;
  }
  __syncthreads();

  // ---------------- Phase 2: weights out + AV accumulate ------------------
  {
    const int rgs = wave >> 1;            // score row group for this wave
    const int qh  = wave & 1;             // which 16-col half of the 32-col step
    float invl[8];
    #pragma unroll
    for (int r = 0; r < 8; ++r) invl[r] = stats[rgs * 16 + hb * 8 + r];

    v8f acc[4][2];
    #pragma unroll
    for (int rg = 0; rg < 4; ++rg) {
      acc[rg][0] = vzero;
      acc[rg][1] = vzero;
    }

    const __bf16* ctxPb = ctxP + (size_t)b * 8 * LEN * 32;
    const __bf16* ctxbb = ctxb + (size_t)b * DIM * LEN;
    const int dw = wave * 32;             // this wave's 32 output d-columns
    float* wrow = weights + ((size_t)b * LEN + o0 + rgs * 16 + hb * 8) * LEN;

    #pragma unroll 1
    for (int jt2 = 0; jt2 < 128; ++jt2) {
      const int q0 = jt2 * 32;
      const int qs = q0 + qh * 16;

      // scores tile (rgs, qs..qs+15)
      v8f sacc = vzero;
      #pragma unroll
      for (int kc = 0; kc < 8; ++kc) {
        const __bf16* bp = ctxPb + ((size_t)kc * LEN + qs + nlo) * 32 + hb * 8;
        v16bf bf = load_frag2(bp, bp + 16);
        const __bf16* ap = qpb + (rgs * 16 + nlo) * QSTR + kc * 32 + hb * 8;
        v16bf af = load_frag2(ap, ap + 16);
        sacc = wmma_bf16(af, bf, sacc);
      }
      // normalized softmax: write weights, stage bf16 p for AV A-fragments
      #pragma unroll
      for (int r = 0; r < 8; ++r) {
        float p = __expf(sacc[r]) * invl[r];
        wrow[(size_t)r * LEN + qs + nlo] = p;
        pbuf[(rgs * 16 + hb * 8 + r) * PBSTR + qh * 16 + nlo] = (__bf16)p;
      }
      __syncthreads();

      // AV: K = q0..q0+31, 32 d-columns per wave; B shared across row groups
      const __bf16* vp0 = ctxbb + (size_t)(dw + nlo) * LEN + q0 + hb * 8;
      const __bf16* vp1 = ctxbb + (size_t)(dw + 16 + nlo) * LEN + q0 + hb * 8;
      v16bf bf0 = load_frag2(vp0, vp0 + 16);
      v16bf bf1 = load_frag2(vp1, vp1 + 16);
      __builtin_prefetch(vp0 + 32, 0, 1);     // next K-step -> global_prefetch
      #pragma unroll
      for (int rg = 0; rg < 4; ++rg) {
        const __bf16* ap = pbuf + (rg * 16 + nlo) * PBSTR + hb * 8;
        v16bf af = load_frag2(ap, ap + 16);
        acc[rg][0] = wmma_bf16(af, bf0, acc[rg][0]);
        acc[rg][1] = wmma_bf16(af, bf1, acc[rg][1]);
      }
      __syncthreads();
    }

    float* orow = out + ((size_t)b * LEN + o0) * DIM;
    #pragma unroll
    for (int rg = 0; rg < 4; ++rg)
      #pragma unroll
      for (int nt = 0; nt < 2; ++nt)
        #pragma unroll
        for (int r = 0; r < 8; ++r)
          orow[(size_t)(rg * 16 + hb * 8 + r) * DIM + dw + nt * 16 + nlo] =
              acc[rg][nt][r];
  }
}

extern "C" void kernel_launch(void* const* d_in, const int* in_sizes, int n_in,
                              void* d_out, int out_size, void* d_ws,
                              size_t ws_size, hipStream_t stream) {
  const float* query   = (const float*)d_in[0];
  const float* context = (const float*)d_in[1];
  const float* W_in    = (const float*)d_in[2];
  const float* b_in    = (const float*)d_in[3];

  float* out     = (float*)d_out;                          // [B][L][D]
  float* weights = out + (size_t)BATCH * LEN * DIM;        // [B][L][L]

  __bf16* ctxb = (__bf16*)d_ws;                            // 8.4 MB
  __bf16* ctxP = ctxb + (size_t)BATCH * DIM * LEN;         // 8.4 MB
  __bf16* qT   = ctxP + (size_t)BATCH * DIM * LEN;         // 8.4 MB
  __bf16* Wb   = qT + (size_t)BATCH * DIM * LEN;           // 128 KB

  const int prep_blocks = (BATCH * DIM * LEN) / 256;       // 16384
  prep_all<<<prep_blocks, 256, 0, stream>>>(context, query, W_in, ctxb, ctxP,
                                            qT, Wb);

  attn_kernel<<<BATCH * (LEN / 64), 256, 0, stream>>>(qT, Wb, b_in, ctxb, ctxP,
                                                      out, weights);
}
